// TransformerBlock_85547158602177
// MI455X (gfx1250) — matrix operand order repaired and checked by reading, not compiled, not run on hardware
//
#include <hip/hip_runtime.h>
#include <hip/hip_bf16.h>
#include <math.h>

// ---------------------------------------------------------------------------
// TransformerBlock for MI455X (gfx1250): bf16 WMMA GEMMs (fp32 accumulate),
// transposed L2-resident weights, double-buffered TDM (tensor_load_to_lds)
// tile staging overlapped with WMMA via TENSORcnt, fused LN / online-softmax
// attention / exact-GELU MLP.
// ---------------------------------------------------------------------------

typedef unsigned short     u16;
typedef unsigned int       u32;
typedef unsigned long long u64;
typedef __attribute__((ext_vector_type(16))) __bf16 bf16x16;
typedef __attribute__((ext_vector_type(8)))  float  f32x8;
typedef __attribute__((ext_vector_type(4)))  unsigned int u32x4;
typedef __attribute__((ext_vector_type(8)))  int i32x8;
typedef __attribute__((ext_vector_type(4)))  int i32x4;

union AFrag { u32 u[8]; u16 h[16]; bf16x16 v; uint4 q[2]; };
union Acc   { f32x8 v; float f[8]; };

#define SEQ  1024
#define CDIM 1024
#define HIDD 4096
#define ROWS 8192   // B*N

#if defined(__HIP_DEVICE_COMPILE__) && __has_builtin(__builtin_amdgcn_tensor_load_to_lds)
#define USE_TDM 1
#else
#define USE_TDM 0
#endif

__device__ __forceinline__ u16 f2bf(float f) {
  u32 x = __float_as_uint(f);
  return (u16)((x + 0x7FFFu + ((x >> 16) & 1u)) >> 16);
}

__device__ __forceinline__ float waveRedSum(float v) {
  #pragma unroll
  for (int m = 16; m >= 1; m >>= 1) v += __shfl_xor(v, m, 32);
  return v;
}

__device__ __forceinline__ void blockRed2(float& a, float& b, float* rs, float* rq) {
  __syncthreads();
  a = waveRedSum(a);
  b = waveRedSum(b);
  int lane = threadIdx.x & 31, wid = threadIdx.x >> 5;
  if (lane == 0) { rs[wid] = a; rq[wid] = b; }
  __syncthreads();
  if (threadIdx.x == 0) {
    float sa = 0.f, sb = 0.f;
    #pragma unroll
    for (int i = 0; i < 8; i++) { sa += rs[i]; sb += rq[i]; }
    rs[0] = sa; rq[0] = sb;
  }
  __syncthreads();
  a = rs[0]; b = rq[0];
}

#if USE_TDM
// TDM 2-D tile load: global (row-major, strideElems u16/row) -> LDS
// (contiguous tileW x tileH u16). D# per CDNA5 ISA 8.3/8.4: count=1,
// type=2(image), data_size=1(2 bytes), huge tensor dims (tiles in-bounds).
__device__ __forceinline__ void tdm_load_2d(u32 ldsOff, const void* gptr,
                                            u32 tileW, u32 tileH, u32 strideElems) {
  u64 ga = (u64)(uintptr_t)gptr;
  u32x4 g0;
  g0.x = 1u;                                          // count=1 (valid user D#)
  g0.y = ldsOff;                                      // lds_addr (bytes)
  g0.z = (u32)ga;                                     // global_addr[31:0]
  g0.w = (u32)((ga >> 32) & 0x01FFFFFFu) | (2u << 30);// addr[56:32] | type=2
  i32x8 g1;
  const u32 d0 = 0x7FFFFFFFu, d1 = 0x7FFFFFFFu;       // tensor_dim0/1 (huge)
  g1[0] = (int)(1u << 16);                            // data_size=1 (2B), mask=0
  g1[1] = (int)((d0 & 0xFFFFu) << 16);                // tensor_dim0[15:0]
  g1[2] = (int)(((d0 >> 16) & 0xFFFFu) | ((d1 & 0xFFFFu) << 16));
  g1[3] = (int)(((d1 >> 16) & 0xFFFFu) | (tileW << 16)); // tile_dim0
  g1[4] = (int)tileH;                                 // tile_dim1 (tile_dim2=0)
  g1[5] = (int)strideElems;                           // tensor_dim0_stride[31:0]
  g1[6] = 0;
  g1[7] = 0;
  i32x4 z4 = {0, 0, 0, 0};
#if defined(__clang_major__) && (__clang_major__ >= 23)
  i32x8 z8 = {0, 0, 0, 0, 0, 0, 0, 0};
  __builtin_amdgcn_tensor_load_to_lds(g0, g1, z4, z4, z8, 0);
#else
  __builtin_amdgcn_tensor_load_to_lds(g0, g1, z4, z4, 0);
#endif
}
#endif

// ---------------------------------------------------------------------------
// Transposing fp32 -> bf16 weight conversion: dst[n*K + k] = bf16(src[k*N + n])
// ---------------------------------------------------------------------------
__global__ void __launch_bounds__(256) transpose_f2bf_kernel(
    const float* __restrict__ src, u16* __restrict__ dst, int K, int N) {
  __shared__ float t[32][33];
  int kb = blockIdx.x * 32, nb = blockIdx.y * 32;
  int tx = threadIdx.x & 31, ty = threadIdx.x >> 5;   // 32 x 8
  #pragma unroll
  for (int i = ty; i < 32; i += 8)
    t[i][tx] = src[(size_t)(kb + i) * N + nb + tx];
  __syncthreads();
  #pragma unroll
  for (int i = ty; i < 32; i += 8)
    dst[(size_t)(nb + i) * K + kb + tx] = f2bf(t[tx][i]);
}

// ---------------------------------------------------------------------------
// Fused: xpe = x + pos ; y = ln1(xpe) ; xr = xpe + y ; h = ln2(xr)
// ---------------------------------------------------------------------------
__global__ void __launch_bounds__(256) ln_prep_kernel(
    const float* __restrict__ x,  const float* __restrict__ pos,
    const float* __restrict__ g1, const float* __restrict__ b1,
    const float* __restrict__ g2, const float* __restrict__ b2,
    u16* __restrict__ yb, u16* __restrict__ hb, float* __restrict__ xr) {
  __shared__ float rs[8], rq[8];
  int row = blockIdx.x;
  int tid = threadIdx.x;
  const float* xrow = x   + (size_t)row * CDIM;
  const float* prow = pos + (size_t)(row & (SEQ - 1)) * CDIM;

  float v[4];
  float s = 0.f, q = 0.f;
  #pragma unroll
  for (int i = 0; i < 4; i++) {
    int c = tid + i * 256;
    float t = xrow[c] + prow[c];
    v[i] = t; s += t; q += t * t;
  }
  blockRed2(s, q, rs, rq);
  float mean = s * (1.0f / CDIM);
  float inv  = rsqrtf(q * (1.0f / CDIM) - mean * mean + 1e-5f);

  float w[4];
  float s2 = 0.f, q2 = 0.f;
  #pragma unroll
  for (int i = 0; i < 4; i++) {
    int c = tid + i * 256;
    float yv = (v[i] - mean) * inv * g1[c] + b1[c];
    yb[(size_t)row * CDIM + c] = f2bf(yv);
    float t = v[i] + yv;
    xr[(size_t)row * CDIM + c] = t;
    w[i] = t; s2 += t; q2 += t * t;
  }
  blockRed2(s2, q2, rs, rq);
  float mean2 = s2 * (1.0f / CDIM);
  float inv2  = rsqrtf(q2 * (1.0f / CDIM) - mean2 * mean2 + 1e-5f);
  #pragma unroll
  for (int i = 0; i < 4; i++) {
    int c = tid + i * 256;
    hb[(size_t)row * CDIM + c] = f2bf((w[i] - mean2) * inv2 * g2[c] + b2[c]);
  }
}

// One K-step of the wave's 32x16 output: B frag from transposed weights
// (contiguous b128 global loads), two A frags from LDS, two WMMAs.
__device__ __forceinline__ void gemm_step(const u16* sAbase, const u16* wrow,
                                          int kk, int lane, int wm,
                                          Acc& accT, Acc& accB) {
  AFrag b;
  const uint4* bp = (const uint4*)(wrow + kk);
  b.q[0] = bp[0];
  b.q[1] = bp[1];
  AFrag aT, aB;
  const uint4* rT = (const uint4*)(sAbase + (((wm << 5) + (lane & 15)) << 5));
  const uint4* rB = (const uint4*)(sAbase + (((wm << 5) + 16 + (lane & 15)) << 5));
  int q0 = (lane >> 4) << 1, q1 = ((lane >> 4) << 1) + 1;
  aT.q[0] = rT[q0]; aT.q[1] = rT[q1];
  aB.q[0] = rB[q0]; aB.q[1] = rB[q1];
  accT.v = __builtin_amdgcn_wmma_f32_16x16x32_bf16(false, aT.v, false, b.v,
                                                   (short)0, accT.v, false, false);
  accB.v = __builtin_amdgcn_wmma_f32_16x16x32_bf16(false, aB.v, false, b.v,
                                                   (short)0, accB.v, false, false);
}

// ---------------------------------------------------------------------------
// bf16 GEMM: out = act(A[M,K] @ W[K,N] + bias) (+resid), W given TRANSPOSED
// as Wt[N,K]. 64x64 block tile, 8 waves, wave -> 32(M)x16(N) via two
// v_wmma_f32_16x16x32_bf16 per K-step. A tiles staged by the TDM into a
// double-buffered LDS tile so the DMA overlaps the WMMAs (TENSORcnt<=1).
// ---------------------------------------------------------------------------
__global__ void __launch_bounds__(256) gemm_bf16_kernel(
    const u16* __restrict__ A, const u16* __restrict__ Wt,
    const float* __restrict__ bias,
    float* __restrict__ outF, u16* __restrict__ outB,
    const float* __restrict__ resid,
    int M, int N, int K, int doGelu) {
  __shared__ u16 sA[2][64][32];
  int tid = threadIdx.x, lane = tid & 31, wave = tid >> 5;
  int tileN = blockIdx.x * 64, tileM = blockIdx.y * 64;
  int wm = wave >> 2, wn = wave & 3;
  (void)M;

  Acc accT, accB;
  #pragma unroll
  for (int i = 0; i < 8; i++) { accT.f[i] = 0.f; accB.f[i] = 0.f; }

  // per-lane row of transposed weights supplying the whole B fragment
  const u16* wrow = Wt + (size_t)(tileN + (wn << 4) + (lane & 15)) * K
                       + ((lane >> 4) << 4);
  const u16* Abase = A + (size_t)tileM * K;

#if USE_TDM
  int steps = K >> 5;
  if (wave == 0)
    tdm_load_2d((u32)(uintptr_t)&sA[0][0][0], Abase, 32u, 64u, (u32)K);
  for (int i = 0; i < steps; i++) {
    int cur = i & 1;
    int kk = i << 5;
    if (wave == 0) {
      if (i + 1 < steps) {
        // issue next tile into the alternate buffer, then wait only for the
        // *current* tile (oldest outstanding; TDM completes in order)
        tdm_load_2d((u32)(uintptr_t)&sA[cur ^ 1][0][0],
                    Abase + kk + 32, 32u, 64u, (u32)K);
        __builtin_amdgcn_s_wait_tensorcnt(1);
      } else {
        __builtin_amdgcn_s_wait_tensorcnt(0);
      }
    }
    __syncthreads();                                  // sA[cur] visible to all
    if (kk + 32 < K) __builtin_prefetch(wrow + kk + 32, 0, 1);
    gemm_step(&sA[cur][0][0], wrow, kk, lane, wm, accT, accB);
    __syncthreads();                                  // sA[cur] fully consumed
  }
#else
  int arow = tid >> 2, acol = (tid & 3) * 8;          // cooperative A staging
  for (int kk = 0; kk < K; kk += 32) {
    __syncthreads();
    *(uint4*)&sA[0][arow][acol] =
        *(const uint4*)&Abase[(size_t)arow * K + kk + acol];
    __syncthreads();
    if (kk + 32 < K) __builtin_prefetch(wrow + kk + 32, 0, 1);
    gemm_step(&sA[0][0][0], wrow, kk, lane, wm, accT, accB);
  }
#endif

  int gcol = tileN + (wn << 4) + (lane & 15);
  float bv = bias ? bias[gcol] : 0.0f;
  #pragma unroll
  for (int half = 0; half < 2; half++) {
    Acc* acc = half ? &accB : &accT;
    #pragma unroll
    for (int r = 0; r < 8; r++) {
      int grow = tileM + (wm << 5) + (half << 4) + ((lane >> 4) << 3) + r;
      float v0 = acc->f[r] + bv;
      if (doGelu) v0 = 0.5f * v0 * (1.0f + erff(v0 * 0.70710678118654752f));
      size_t idx = (size_t)grow * N + gcol;
      if (resid) v0 += resid[idx];
      if (outF) outF[idx] = v0;
      if (outB) outB[idx] = f2bf(v0);
    }
  }
}

// ---------------------------------------------------------------------------
// Attention: one wave per (b, h, 16-query tile). Online softmax; raw logits
// streamed to the attn output region and normalized in place at the end.
// V-chunk TDM is issued at the top of each iteration so the DMA overlaps the
// whole QK^T + softmax phase; TENSORcnt is drained just before the PV WMMAs.
// ---------------------------------------------------------------------------
__global__ void __launch_bounds__(32) attn_kernel(
    const u16* __restrict__ qb, const u16* __restrict__ kb,
    const u16* __restrict__ vb,
    float* __restrict__ attnOut, u16* __restrict__ aob) {
  __shared__ u16 sP[16][32];
  __shared__ u16 sV[32][64];
  __shared__ float sM[16], sL[16];

  int lane = threadIdx.x;
  int qt = blockIdx.x;              // 0..63  (query tile)
  int bh = blockIdx.y;              // 0..127 (b*16 + h)
  int b = bh >> 4, h = bh & 15;
  int qbase = qt << 4;
  size_t rowBase = (size_t)b * SEQ;
  int hcol = h << 6;

  // Q fragments for d=0..31 and d=32..63 (loaded once)
  AFrag qa[2];
  {
    const uint4* qrow =
        (const uint4*)(qb + (rowBase + qbase + (lane & 15)) * CDIM + hcol);
    #pragma unroll
    for (int dh = 0; dh < 2; dh++) {
      qa[dh].q[0] = qrow[(dh << 2) + ((lane >> 4) << 1)];
      qa[dh].q[1] = qrow[(dh << 2) + ((lane >> 4) << 1) + 1];
    }
  }

  float mr[8], lr[8];
  Acc accO[4];
  #pragma unroll
  for (int r = 0; r < 8; r++) { mr[r] = -1e30f; lr[r] = 0.f; }
  #pragma unroll
  for (int dt = 0; dt < 4; dt++)
    #pragma unroll
    for (int r = 0; r < 8; r++) accO[dt].f[r] = 0.f;

  float* attnRow = attnOut + ((size_t)bh * SEQ + qbase) * SEQ;

  for (int jc = 0; jc < 32; jc++) {       // 32-key chunks
    int jbase = jc << 5;

    __syncthreads();  // retire previous iteration's sP/sV readers
#if USE_TDM
    // issue V-chunk DMA early: overlaps QK^T + softmax below
    tdm_load_2d((u32)(uintptr_t)&sV[0][0],
                vb + (rowBase + jbase) * CDIM + hcol, 64u, 32u, (u32)CDIM);
#endif

    // S = (Q K^T) / sqrt(d), two 16-key subtiles; K frags contiguous b128
    Acc s[2];
    #pragma unroll
    for (int st = 0; st < 2; st++) {
      #pragma unroll
      for (int r = 0; r < 8; r++) s[st].f[r] = 0.f;
      const u16* krow =
          kb + (rowBase + jbase + (st << 4) + (lane & 15)) * CDIM + hcol;
      #pragma unroll
      for (int dh = 0; dh < 2; dh++) {
        AFrag bK;
        const uint4* kp = (const uint4*)(krow + (dh << 5) + ((lane >> 4) << 4));
        bK.q[0] = kp[0];
        bK.q[1] = kp[1];
        s[st].v = __builtin_amdgcn_wmma_f32_16x16x32_bf16(
            false, qa[dh].v, false, bK.v, (short)0, s[st].v, false, false);
      }
      #pragma unroll
      for (int r = 0; r < 8; r++) s[st].f[r] *= 0.125f;
    }

    // stream raw logits (normalized in place later)
    #pragma unroll
    for (int st = 0; st < 2; st++)
      #pragma unroll
      for (int r = 0; r < 8; r++) {
        int rloc = ((lane >> 4) << 3) + r;
        attnRow[(size_t)rloc * SEQ + jbase + (st << 4) + (lane & 15)] = s[st].f[r];
      }

    // online softmax update
    float scale[8];
    #pragma unroll
    for (int r = 0; r < 8; r++) {
      float cm = fmaxf(s[0].f[r], s[1].f[r]);
      #pragma unroll
      for (int msk = 1; msk < 16; msk <<= 1) cm = fmaxf(cm, __shfl_xor(cm, msk, 32));
      float mn = fmaxf(mr[r], cm);
      float sc = __expf(mr[r] - mn);
      float p0 = __expf(s[0].f[r] - mn);
      float p1 = __expf(s[1].f[r] - mn);
      float rsum = p0 + p1;
      #pragma unroll
      for (int msk = 1; msk < 16; msk <<= 1) rsum += __shfl_xor(rsum, msk, 32);
      lr[r] = lr[r] * sc + rsum;
      mr[r] = mn;
      scale[r] = sc;
      s[0].f[r] = p0; s[1].f[r] = p1;
    }
    #pragma unroll
    for (int dt = 0; dt < 4; dt++)
      #pragma unroll
      for (int r = 0; r < 8; r++) accO[dt].f[r] *= scale[r];

    // stage P (bf16) into LDS for the A operand of the PV WMMAs
    #pragma unroll
    for (int st = 0; st < 2; st++)
      #pragma unroll
      for (int r = 0; r < 8; r++)
        sP[((lane >> 4) << 3) + r][(st << 4) + (lane & 15)] = f2bf(s[st].f[r]);

#if USE_TDM
    __builtin_amdgcn_s_wait_tensorcnt(0);   // V chunk landed in sV
#else
    {
      const uint4* vr = (const uint4*)(vb + (rowBase + jbase + lane) * CDIM + hcol);
      uint4* dr = (uint4*)&sV[lane][0];
      #pragma unroll
      for (int i = 0; i < 8; i++) dr[i] = vr[i];
    }
#endif
    __syncthreads();

    AFrag aP;
    {
      const uint4* pr = (const uint4*)&sP[lane & 15][0];
      aP.q[0] = pr[((lane >> 4) << 1)];
      aP.q[1] = pr[((lane >> 4) << 1) + 1];
    }
    #pragma unroll
    for (int dt = 0; dt < 4; dt++) {
      AFrag bV;
      #pragma unroll
      for (int k2 = 0; k2 < 16; k2++)
        bV.h[k2] = sV[((lane >> 4) << 4) + k2][(dt << 4) + (lane & 15)];
      accO[dt].v = __builtin_amdgcn_wmma_f32_16x16x32_bf16(
          false, aP.v, false, bV.v, (short)0, accO[dt].v, false, false);
    }
  }

  // head outputs (bf16) for the projection GEMM
  #pragma unroll
  for (int dt = 0; dt < 4; dt++)
    #pragma unroll
    for (int r = 0; r < 8; r++) {
      int rloc = ((lane >> 4) << 3) + r;
      float o = accO[dt].f[r] / lr[r];
      aob[(rowBase + qbase + rloc) * CDIM + hcol + (dt << 4) + (lane & 15)] = f2bf(o);
    }

  // share per-row stats, normalize logits -> probabilities in place
  if ((lane & 15) == 0)
    #pragma unroll
    for (int r = 0; r < 8; r++) {
      int rloc = ((lane >> 4) << 3) + r;
      sM[rloc] = mr[r]; sL[rloc] = lr[r];
    }
  __syncthreads();
  for (int idx = lane; idx < 16 * SEQ; idx += 32) {
    int rloc = idx >> 10, key = idx & (SEQ - 1);
    float lg = attnRow[(size_t)rloc * SEQ + key];
    attnRow[(size_t)rloc * SEQ + key] = __expf(lg - sM[rloc]) / sL[rloc];
  }
}

// ---------------------------------------------------------------------------
// Host launcher
// ---------------------------------------------------------------------------
extern "C" void kernel_launch(void* const* d_in, const int* in_sizes, int n_in,
                              void* d_out, int out_size, void* d_ws, size_t ws_size,
                              hipStream_t stream) {
  (void)in_sizes; (void)n_in; (void)out_size; (void)ws_size;
  const float* x      = (const float*)d_in[0];
  const float* pos    = (const float*)d_in[1];
  const float* wq     = (const float*)d_in[2];
  const float* wk     = (const float*)d_in[3];
  const float* wv     = (const float*)d_in[4];
  const float* proj_w = (const float*)d_in[5];
  const float* proj_b = (const float*)d_in[6];
  const float* ln1_g  = (const float*)d_in[7];
  const float* ln1_b  = (const float*)d_in[8];
  const float* ln2_g  = (const float*)d_in[9];
  const float* ln2_b  = (const float*)d_in[10];
  const float* fc1_w  = (const float*)d_in[11];
  const float* fc1_b  = (const float*)d_in[12];
  const float* fc2_w  = (const float*)d_in[13];
  const float* fc2_b  = (const float*)d_in[14];

  float* outX    = (float*)d_out;
  float* outAO   = outX  + (size_t)ROWS * CDIM;
  float* outATTN = outAO + (size_t)ROWS * CDIM;

  char* w = (char*)d_ws;
  size_t off = 0;
  #define TAKE(T, count) (T*)(w + off); off = (off + (size_t)(count) * sizeof(T) + 255) & ~(size_t)255
  u16*   yb   = TAKE(u16,   (size_t)ROWS * CDIM);
  u16*   hb   = TAKE(u16,   (size_t)ROWS * CDIM);
  float* xr   = TAKE(float, (size_t)ROWS * CDIM);
  u16*   qbuf = TAKE(u16,   (size_t)ROWS * CDIM);
  u16*   kbuf = TAKE(u16,   (size_t)ROWS * CDIM);
  u16*   vbuf = TAKE(u16,   (size_t)ROWS * CDIM);
  u16*   aob  = TAKE(u16,   (size_t)ROWS * CDIM);
  u16*   z1b  = TAKE(u16,   (size_t)ROWS * HIDD);
  u16*   wqb  = TAKE(u16,   (size_t)CDIM * CDIM);   // transposed bf16 weights
  u16*   wkb  = TAKE(u16,   (size_t)CDIM * CDIM);
  u16*   wvb  = TAKE(u16,   (size_t)CDIM * CDIM);
  u16*   wpb  = TAKE(u16,   (size_t)CDIM * CDIM);
  u16*   f1b  = TAKE(u16,   (size_t)CDIM * HIDD);
  u16*   f2b  = TAKE(u16,   (size_t)HIDD * CDIM);
  #undef TAKE

  // 1) weights -> bf16, transposed to Wt[N][K]
  dim3 t256(256);
  transpose_f2bf_kernel<<<dim3(CDIM / 32, CDIM / 32), t256, 0, stream>>>(wq,     wqb, CDIM, CDIM);
  transpose_f2bf_kernel<<<dim3(CDIM / 32, CDIM / 32), t256, 0, stream>>>(wk,     wkb, CDIM, CDIM);
  transpose_f2bf_kernel<<<dim3(CDIM / 32, CDIM / 32), t256, 0, stream>>>(wv,     wvb, CDIM, CDIM);
  transpose_f2bf_kernel<<<dim3(CDIM / 32, CDIM / 32), t256, 0, stream>>>(proj_w, wpb, CDIM, CDIM);
  transpose_f2bf_kernel<<<dim3(CDIM / 32, HIDD / 32), t256, 0, stream>>>(fc1_w,  f1b, CDIM, HIDD);
  transpose_f2bf_kernel<<<dim3(HIDD / 32, CDIM / 32), t256, 0, stream>>>(fc2_w,  f2b, HIDD, CDIM);

  // 2) fused x+pos / ln1 / residual / ln2
  ln_prep_kernel<<<ROWS, 256, 0, stream>>>(x, pos, ln1_g, ln1_b, ln2_g, ln2_b,
                                           yb, hb, xr);

  // 3) QKV projections (no bias), bf16 outputs
  dim3 gC(CDIM / 64, ROWS / 64);
  gemm_bf16_kernel<<<gC, 256, 0, stream>>>(yb, wqb, nullptr, nullptr, qbuf, nullptr,
                                           ROWS, CDIM, CDIM, 0);
  gemm_bf16_kernel<<<gC, 256, 0, stream>>>(yb, wkb, nullptr, nullptr, kbuf, nullptr,
                                           ROWS, CDIM, CDIM, 0);
  gemm_bf16_kernel<<<gC, 256, 0, stream>>>(yb, wvb, nullptr, nullptr, vbuf, nullptr,
                                           ROWS, CDIM, CDIM, 0);

  // 4) attention: attn probs -> d_out, head outputs (bf16) -> aob
  attn_kernel<<<dim3(SEQ / 16, 8 * 16), 32, 0, stream>>>(qbuf, kbuf, vbuf,
                                                         outATTN, aob);

  // 5) output projection (bias) -> d_out.ao
  gemm_bf16_kernel<<<gC, 256, 0, stream>>>(aob, wpb, proj_b, outAO, nullptr, nullptr,
                                           ROWS, CDIM, CDIM, 0);

  // 6) MLP: fc1 + exact GELU (bf16), fc2 + bias + residual(xr) -> d_out.x
  dim3 gH(HIDD / 64, ROWS / 64);
  gemm_bf16_kernel<<<gH, 256, 0, stream>>>(hb, f1b, fc1_b, nullptr, z1b, nullptr,
                                           ROWS, HIDD, CDIM, 1);
  gemm_bf16_kernel<<<gC, 256, 0, stream>>>(z1b, f2b, fc2_b, outX, nullptr, xr,
                                           ROWS, CDIM, HIDD, 0);
}
